// get_model_61263413510337
// MI455X (gfx1250) — matrix operand order repaired and checked by reading, not compiled, not run on hardware
//
#include <hip/hip_runtime.h>
#include <hip/hip_bf16.h>
#include <math.h>

typedef __attribute__((ext_vector_type(16))) _Float16 v16h;
typedef __attribute__((ext_vector_type(8)))  _Float16 v8h;
typedef __attribute__((ext_vector_type(8)))  float    v8f;

#define BB 8  // batch size (fixed by reference)

static inline int pad16(int x) { return (x + 15) & ~15; }
static inline int pad32(int x) { return (x + 31) & ~31; }

// ---------------------------------------------------------------------------
// Weight prep: f32 (cout,cin) -> f16 (coutP,cinP) zero-padded; fold BN affine
// y = relu((x@W^T + b) * s * g + beta),  s = 1/sqrt(1+eps)
//   => y = relu(acc * alpha[o] + bias[o]),  alpha = s*g, bias = b*s*g + beta
// ---------------------------------------------------------------------------
__global__ void prep_layer_kernel(const float* __restrict__ w, const float* __restrict__ b,
                                  const float* __restrict__ g, const float* __restrict__ beta,
                                  int cout, int cin, int coutP, int cinP, int has_bn,
                                  _Float16* __restrict__ wh, float* __restrict__ alpha,
                                  float* __restrict__ bias) {
  int i = blockIdx.x * blockDim.x + threadIdx.x;
  if (i < coutP * cinP) {
    int o = i / cinP, c = i % cinP;
    float v = (o < cout && c < cin) ? w[(size_t)o * cin + c] : 0.f;
    wh[i] = (_Float16)v;
  }
  if (i < coutP) {
    float a = 0.f, bi = 0.f;
    if (i < cout) {
      float s = rsqrtf(1.f + 1e-5f);
      if (has_bn) { a = s * g[i]; bi = b[i] * s * g[i] + beta[i]; }
      else        { a = 1.f;      bi = b[i]; }
    }
    alpha[i] = a; bias[i] = bi;
  }
}

__global__ void zero_f32_kernel(float* __restrict__ p, int n) {
  int t = blockIdx.x * blockDim.x + threadIdx.x;
  if (t < n) p[t] = 0.f;
}

// ---------------------------------------------------------------------------
// Farthest point sampling: one block per batch, dist array in LDS.
// ---------------------------------------------------------------------------
__global__ void fps_kernel(const float* __restrict__ xyz, int N, int npoint,
                           int* __restrict__ out_idx) {
  __shared__ float dist[2048];
  __shared__ float rmax[256];
  __shared__ int   rmaxi[256];
  int b = blockIdx.x;
  const float* p = xyz + (size_t)b * N * 3;
  for (int i = threadIdx.x; i < N; i += blockDim.x) dist[i] = 1e10f;
  __syncthreads();
  int far = 0;
  for (int it = 0; it < npoint; ++it) {
    if (threadIdx.x == 0) out_idx[b * npoint + it] = far;
    float cx = p[far * 3 + 0], cy = p[far * 3 + 1], cz = p[far * 3 + 2];
    float best = -1.f; int besti = 0;
    for (int i = threadIdx.x; i < N; i += blockDim.x) {
      float dx = p[i * 3] - cx, dy = p[i * 3 + 1] - cy, dz = p[i * 3 + 2] - cz;
      float d = dx * dx + dy * dy + dz * dz;
      float dd = fminf(dist[i], d);
      dist[i] = dd;
      if (dd > best) { best = dd; besti = i; }
    }
    rmax[threadIdx.x] = best; rmaxi[threadIdx.x] = besti;
    __syncthreads();
    for (int s = 128; s > 0; s >>= 1) {
      if (threadIdx.x < s && rmax[threadIdx.x + s] > rmax[threadIdx.x]) {
        rmax[threadIdx.x] = rmax[threadIdx.x + s];
        rmaxi[threadIdx.x] = rmaxi[threadIdx.x + s];
      }
      __syncthreads();
    }
    far = rmaxi[0];
    __syncthreads();
  }
}

__global__ void gather_xyz_kernel(const float* __restrict__ src, const int* __restrict__ idx,
                                  int N, int S, float* __restrict__ dst) {
  int t = blockIdx.x * blockDim.x + threadIdx.x;
  if (t >= BB * S * 3) return;
  int c = t % 3, s = (t / 3) % S, b = t / (3 * S);
  dst[t] = src[((size_t)b * N + idx[b * S + s]) * 3 + c];
}

// ---------------------------------------------------------------------------
// Ball query: first K indices (ascending) within r2; pad with first hit.
// ---------------------------------------------------------------------------
__global__ void ball_query_kernel(const float* __restrict__ xyz, const float* __restrict__ new_xyz,
                                  int N, int S, float r2, int K, int* __restrict__ idx) {
  int t = blockIdx.x * blockDim.x + threadIdx.x;
  if (t >= BB * S) return;
  int b = t / S;
  const float* p = xyz + (size_t)b * N * 3;
  float qx = new_xyz[(size_t)t * 3 + 0];
  float qy = new_xyz[(size_t)t * 3 + 1];
  float qz = new_xyz[(size_t)t * 3 + 2];
  int* o = idx + (size_t)t * K;
  int cnt = 0;
  for (int j = 0; j < N && cnt < K; ++j) {
    float dx = p[j * 3] - qx, dy = p[j * 3 + 1] - qy, dz = p[j * 3 + 2] - qz;
    if (dx * dx + dy * dy + dz * dz <= r2) o[cnt++] = j;
  }
  int f = (cnt > 0) ? o[0] : 0;
  for (; cnt < K; ++cnt) o[cnt] = f;
}

// ---------------------------------------------------------------------------
// Build grouped f16 rows: [feat(C) | rel_xyz(3) | zero-pad] per (b,s,k).
// ---------------------------------------------------------------------------
__global__ void group_rows_kernel(const float* __restrict__ feat, int C,
                                  const float* __restrict__ xyz, const float* __restrict__ new_xyz,
                                  const int* __restrict__ idx,
                                  int N, int S, int K, int cinP, _Float16* __restrict__ X) {
  int row = blockIdx.x;                 // BB*S*K rows
  int s = (row / K) % S, b = row / (K * S);
  int j = idx[row];
  const float* f = feat ? feat + ((size_t)b * N + j) * C : nullptr;
  float rx = xyz[((size_t)b * N + j) * 3 + 0] - new_xyz[((size_t)b * S + s) * 3 + 0];
  float ry = xyz[((size_t)b * N + j) * 3 + 1] - new_xyz[((size_t)b * S + s) * 3 + 1];
  float rz = xyz[((size_t)b * N + j) * 3 + 2] - new_xyz[((size_t)b * S + s) * 3 + 2];
  _Float16* o = X + (size_t)row * cinP;
  for (int c = threadIdx.x; c < cinP; c += blockDim.x) {
    float v = 0.f;
    if (c < C)          v = f[c];
    else if (c == C)    v = rx;
    else if (c == C + 1) v = ry;
    else if (c == C + 2) v = rz;
    o[c] = (_Float16)v;
  }
}

// ---------------------------------------------------------------------------
// Core fused GEMM: Y = act(X @ W^T * alpha + bias).
// One wave = one 16-row M tile x NT 16-col N tiles (register blocking along N:
// A fragment loaded once per k-step and reused by NT back-to-back WMMAs).
// Optional fused group-max-pool epilogue: every 16-row M tile lies inside one
// pooling group (Kgroup is a multiple of 16), so reduce 8 rows/lane with fmax,
// fold the two half-wave row halves with shfl_xor(16), and emit one integer
// atomicMax per column per tile (valid: post-ReLU values >= 0; deterministic).
// A/B use contiguous k (half hi: k = 16*hi + 0..15); D per CDNA5 ISA 7.12.2 for v_wmma_f32_16x16x32_f16.
// ---------------------------------------------------------------------------
template <int NT>
__global__ __launch_bounds__(256) void wmma_mlp_kernel(
    const _Float16* __restrict__ X, int ldx,
    const _Float16* __restrict__ W, int ldw,
    const float* __restrict__ alpha, const float* __restrict__ bias,
    int M, int Kdim, int Np,
    _Float16* __restrict__ Yh, int ldy,
    float* __restrict__ Yf, int ldyf, int do_relu,
    float* __restrict__ poolDst, int ldpool, int coff, int Kgroup) {
  int wave = threadIdx.x >> 5;
  int lane = threadIdx.x & 31;
  int mtiles = M >> 4;
  int nblocks = Np / (16 * NT);
  int t = blockIdx.x * (blockDim.x >> 5) + wave;
  if (t >= mtiles * nblocks) return;          // wave-uniform: EXEC stays all-ones
  int mt = t % mtiles, nb = t / mtiles;
  int l16 = lane & 15, hi = lane >> 4;
  const _Float16* xr = X + (size_t)(mt * 16 + l16) * ldx + hi * 16;
  const _Float16* wr = W + (size_t)(nb * NT * 16 + l16) * ldw + hi * 16;
  v8f acc[NT];
#pragma unroll
  for (int j = 0; j < NT; ++j) acc[j] = (v8f){};
  for (int k = 0; k < Kdim; k += 32) {
    __builtin_prefetch(xr + k + 256, 0, 1);
    v8h alo = *(const v8h*)(xr + k);          // K = k + hi*16 + [0..7]
    v8h ahi = *(const v8h*)(xr + k + 8);      // K = k + hi*16 + 8 + [0..7]
    v16h a = __builtin_shufflevector(alo, ahi, 0, 1, 2, 3, 4, 5, 6, 7,
                                     8, 9, 10, 11, 12, 13, 14, 15);
#pragma unroll
    for (int j = 0; j < NT; ++j) {
      v16h bm = *(const v16h*)(wr + (size_t)j * 16 * ldw + k);  // K = k+hi*16+[0..15]
      acc[j] = __builtin_amdgcn_wmma_f32_16x16x32_f16(false, a, false, bm,
                                                      (short)0, acc[j], false, false);
    }
  }
  if (poolDst) {
    int grp = (mt * 16) / Kgroup;             // tile fully inside one group
#pragma unroll
    for (int j = 0; j < NT; ++j) {
      int col = (nb * NT + j) * 16 + l16;
      float al = alpha[col], bi = bias[col];
      float m = 0.f;                          // ReLU folded into the max
#pragma unroll
      for (int r = 0; r < 8; ++r) m = fmaxf(m, acc[j][r] * al + bi);
      m = fmaxf(m, __shfl_xor(m, 16));        // combine row halves (same col)
      if (hi == 0)
        atomicMax((int*)(poolDst + (size_t)grp * ldpool + coff + col),
                  __float_as_int(m));
    }
  } else {
#pragma unroll
    for (int j = 0; j < NT; ++j) {
      int col = (nb * NT + j) * 16 + l16;
      float al = alpha[col], bi = bias[col];
#pragma unroll
      for (int r = 0; r < 8; ++r) {
        int rowm = mt * 16 + r + hi * 8;
        float y = acc[j][r] * al + bi;
        if (do_relu) y = fmaxf(y, 0.f);
        if (Yh) Yh[(size_t)rowm * ldy + col] = (_Float16)y;
        if (Yf) Yf[(size_t)rowm * ldyf + col] = y;
      }
    }
  }
}

// 3-NN inverse-distance interpolation (feature_propagation).
__global__ void fp_interp_kernel(const float* __restrict__ xyz1, const float* __restrict__ xyz2,
                                 const float* __restrict__ pts2, int N, int S, int C2,
                                 float* __restrict__ interp) {
  int t = blockIdx.x * blockDim.x + threadIdx.x;
  if (t >= BB * N) return;
  int b = t / N;
  const float* q = xyz1 + (size_t)t * 3;
  const float* p = xyz2 + (size_t)b * S * 3;
  float d0 = 1e30f, d1 = 1e30f, d2 = 1e30f; int i0 = 0, i1 = 0, i2 = 0;
  for (int j = 0; j < S; ++j) {
    float dx = p[j * 3] - q[0], dy = p[j * 3 + 1] - q[1], dz = p[j * 3 + 2] - q[2];
    float d = dx * dx + dy * dy + dz * dz;
    if (d < d0)      { d2 = d1; i2 = i1; d1 = d0; i1 = i0; d0 = d; i0 = j; }
    else if (d < d1) { d2 = d1; i2 = i1; d1 = d;  i1 = j; }
    else if (d < d2) { d2 = d;  i2 = j; }
  }
  float w0 = 1.f / (d0 + 1e-8f), w1 = 1.f / (d1 + 1e-8f), w2 = 1.f / (d2 + 1e-8f);
  float ws = w0 + w1 + w2; w0 /= ws; w1 /= ws; w2 /= ws;
  const float* f = pts2 + (size_t)b * S * C2;
  float* o = interp + (size_t)t * C2;
  for (int c = 0; c < C2; ++c)
    o[c] = f[(size_t)i0 * C2 + c] * w0 + f[(size_t)i1 * C2 + c] * w1 + f[(size_t)i2 * C2 + c] * w2;
}

// Generic [a(C1) | b(C2) | pad] -> f16 rows; b optionally batch-broadcast.
__global__ void concat2_f16_kernel(const float* __restrict__ a, int C1,
                                   const float* __restrict__ bsrc, int C2, int bcast,
                                   int Nn, int cinP, _Float16* __restrict__ X) {
  int row = blockIdx.x;                 // BB*Nn rows
  int b = row / Nn;
  const float* ar = a ? a + (size_t)row * C1 : nullptr;
  const float* br = bsrc + (bcast ? (size_t)b * C2 : (size_t)row * C2);
  _Float16* o = X + (size_t)row * cinP;
  for (int c = threadIdx.x; c < cinP; c += blockDim.x) {
    float v = 0.f;
    if (c < C1)            v = ar[c];
    else if (c < C1 + C2)  v = br[c - C1];
    o[c] = (_Float16)v;
  }
}

// fp1 input rows: [cls_onehot(6) | xyz(3) | xyz(3) | interp(128) | pad] -> 160
__global__ void build_fp1_kernel(const float* __restrict__ cls, const float* __restrict__ xyz,
                                 const float* __restrict__ interp, int N, int cinP,
                                 _Float16* __restrict__ X) {
  int row = blockIdx.x;                 // BB*N rows
  int b = row / N;
  const float* q = xyz + (size_t)row * 3;
  const float* it = interp + (size_t)row * 128;
  _Float16* o = X + (size_t)row * cinP;
  for (int c = threadIdx.x; c < cinP; c += blockDim.x) {
    float v = 0.f;
    if (c < 6)        v = cls[b * 6 + c];
    else if (c < 9)   v = q[c - 6];
    else if (c < 12)  v = q[c - 9];
    else if (c < 140) v = it[c - 12];
    o[c] = (_Float16)v;
  }
}

__global__ void log_softmax_kernel(const float* __restrict__ Y, int ld, int C, int M,
                                   float* __restrict__ out) {
  int t = blockIdx.x * blockDim.x + threadIdx.x;
  if (t >= M) return;
  const float* y = Y + (size_t)t * ld;
  float mx = -3.0e38f;
  for (int c = 0; c < C; ++c) mx = fmaxf(mx, y[c]);
  float s = 0.f;
  for (int c = 0; c < C; ++c) s += expf(y[c] - mx);
  float l = logf(s);
  for (int c = 0; c < C; ++c) out[(size_t)t * C + c] = y[c] - mx - l;
}

__global__ void copy_f32_kernel(const float* __restrict__ src, float* __restrict__ dst, int n) {
  int t = blockIdx.x * blockDim.x + threadIdx.x;
  if (t < n) dst[t] = src[t];
}

// ---------------------------------------------------------------------------
// Host orchestration
// ---------------------------------------------------------------------------
struct Layer { const float *w, *b, *g, *beta; int cin, cout; };
struct Prep  { _Float16* wh; float* alpha; float* bias; int cinP, coutP, cout; };

extern "C" void kernel_launch(void* const* d_in, const int* in_sizes, int n_in,
                              void* d_out, int out_size, void* d_ws, size_t ws_size,
                              hipStream_t stream) {
  (void)in_sizes; (void)n_in; (void)out_size; (void)ws_size;
  const float* points = (const float*)d_in[0];     // (8,2048,3)
  const float* cls    = (const float*)d_in[1];     // (8,6)

  // params pytree flattened with sorted dict keys:
  // fp1, fp2, fp3, head, msg1, msg2, ssg; per conv-layer keys: b, beta, g, w
  int cur = 2;
  auto takeLayer = [&](int cin, int cout) -> Layer {
    Layer L;
    L.b = (const float*)d_in[cur++]; L.beta = (const float*)d_in[cur++];
    L.g = (const float*)d_in[cur++]; L.w    = (const float*)d_in[cur++];
    L.cin = cin; L.cout = cout; return L;
  };
  Layer fp1L[2] = { takeLayer(140, 128), takeLayer(128, 128) };
  Layer fp2L[2] = { takeLayer(704, 256), takeLayer(256, 128) };
  Layer fp3L[2] = { takeLayer(1152, 256), takeLayer(256, 256) };
  const float* hb1 = (const float*)d_in[cur++];
  const float* hb2 = (const float*)d_in[cur++];
  const float* hbeta1 = (const float*)d_in[cur++];
  const float* hg1 = (const float*)d_in[cur++];
  const float* hw1 = (const float*)d_in[cur++];
  const float* hw2 = (const float*)d_in[cur++];
  int m1c[3][3] = { {32, 32, 64}, {64, 64, 128}, {128, 128, 256} };
  int m2c[3][3] = { {64, 64, 128}, {128, 128, 256}, {128, 128, 256} };
  Layer msg1L[3][3], msg2L[3][3], ssgL[3];
  for (int i = 0; i < 3; ++i) { int ci = 3;   for (int l = 0; l < 3; ++l) { msg1L[i][l] = takeLayer(ci, m1c[i][l]); ci = m1c[i][l]; } }
  for (int i = 0; i < 3; ++i) { int ci = 451; for (int l = 0; l < 3; ++l) { msg2L[i][l] = takeLayer(ci, m2c[i][l]); ci = m2c[i][l]; } }
  { int ci = 643; int cc[3] = {128, 256, 512}; for (int l = 0; l < 3; ++l) { ssgL[l] = takeLayer(ci, cc[l]); ci = cc[l]; } }

  // --- workspace arena ---
  size_t off = 0;
  auto alloc = [&](size_t bytes) -> void* {
    off = (off + 255) & ~(size_t)255;
    void* p = (char*)d_ws + off;
    off += bytes;
    return p;
  };

  auto prep = [&](const Layer& L, int has_bn) -> Prep {
    Prep P;
    P.cinP = pad32(L.cin); P.coutP = pad16(L.cout); P.cout = L.cout;
    P.wh = (_Float16*)alloc((size_t)P.coutP * P.cinP * 2);
    P.alpha = (float*)alloc((size_t)P.coutP * 4);
    P.bias = (float*)alloc((size_t)P.coutP * 4);
    int total = P.coutP * P.cinP;
    prep_layer_kernel<<<dim3((total + 255) / 256), dim3(256), 0, stream>>>(
        L.w, L.b, L.g, L.beta, L.cout, L.cin, P.coutP, P.cinP, has_bn,
        P.wh, P.alpha, P.bias);
    return P;
  };

  Prep Pm1[3][3], Pm2[3][3], Pssg[3], Pfp3[2], Pfp2[2], Pfp1[2], Ph1, Ph2;
  for (int i = 0; i < 3; ++i) for (int l = 0; l < 3; ++l) Pm1[i][l] = prep(msg1L[i][l], 1);
  for (int i = 0; i < 3; ++i) for (int l = 0; l < 3; ++l) Pm2[i][l] = prep(msg2L[i][l], 1);
  for (int l = 0; l < 3; ++l) Pssg[l] = prep(ssgL[l], 1);
  for (int l = 0; l < 2; ++l) Pfp3[l] = prep(fp3L[l], 1);
  for (int l = 0; l < 2; ++l) Pfp2[l] = prep(fp2L[l], 1);
  for (int l = 0; l < 2; ++l) Pfp1[l] = prep(fp1L[l], 1);
  { Layer h1 = { hw1, hb1, hg1, hbeta1, 128, 128 }; Ph1 = prep(h1, 1); }
  { Layer h2 = { hw2, hb2, nullptr, nullptr, 128, 50 }; Ph2 = prep(h2, 0); }

  // persistent buffers
  int*   fps1   = (int*)alloc(BB * 512 * 4);
  float* l1xyz  = (float*)alloc((size_t)BB * 512 * 3 * 4);
  int*   fps2   = (int*)alloc(BB * 256 * 4);
  float* l2xyz  = (float*)alloc((size_t)BB * 256 * 3 * 4);
  float* l1pts  = (float*)alloc((size_t)BB * 512 * 448 * 4);
  float* l2pts  = (float*)alloc((size_t)BB * 256 * 640 * 4);
  float* l3pts  = (float*)alloc((size_t)BB * 512 * 4);
  float* l2new  = (float*)alloc((size_t)BB * 256 * 256 * 4);
  float* l1new  = (float*)alloc((size_t)BB * 512 * 128 * 4);
  _Float16* l0feat = (_Float16*)alloc((size_t)BB * 2048 * 128 * 2);
  float* logits = (float*)alloc((size_t)BB * 2048 * 64 * 4);
  size_t mark = off;   // everything after this is stage-scoped scratch

  auto runMLPFull = [&](const Prep& P, const _Float16* X, int M,
                        _Float16* Yh, int ldy, float* Yf, int ldyf, int relu,
                        float* poolDst, int ldpool, int coff, int Kgroup) {
    int ntiles = P.coutP / 16;
    int NT = 8; while (ntiles % NT) NT >>= 1;
    int nblocks = ntiles / NT;
    int tiles = (M / 16) * nblocks;
    int blocks = (tiles + 7) / 8;
    switch (NT) {
      case 8:
        wmma_mlp_kernel<8><<<dim3(blocks), dim3(256), 0, stream>>>(
            X, P.cinP, P.wh, P.cinP, P.alpha, P.bias, M, P.cinP, P.coutP,
            Yh, ldy, Yf, ldyf, relu, poolDst, ldpool, coff, Kgroup);
        break;
      case 4:
        wmma_mlp_kernel<4><<<dim3(blocks), dim3(256), 0, stream>>>(
            X, P.cinP, P.wh, P.cinP, P.alpha, P.bias, M, P.cinP, P.coutP,
            Yh, ldy, Yf, ldyf, relu, poolDst, ldpool, coff, Kgroup);
        break;
      case 2:
        wmma_mlp_kernel<2><<<dim3(blocks), dim3(256), 0, stream>>>(
            X, P.cinP, P.wh, P.cinP, P.alpha, P.bias, M, P.cinP, P.coutP,
            Yh, ldy, Yf, ldyf, relu, poolDst, ldpool, coff, Kgroup);
        break;
      default:
        wmma_mlp_kernel<1><<<dim3(blocks), dim3(256), 0, stream>>>(
            X, P.cinP, P.wh, P.cinP, P.alpha, P.bias, M, P.cinP, P.coutP,
            Yh, ldy, Yf, ldyf, relu, poolDst, ldpool, coff, Kgroup);
        break;
    }
  };
  auto runMLP = [&](const Prep& P, const _Float16* X, int M,
                    _Float16* Yh, int ldy, float* Yf, int ldyf, int relu) {
    runMLPFull(P, X, M, Yh, ldy, Yf, ldyf, relu, nullptr, 0, 0, 0);
  };

  auto runScale = [&](const float* xyz, const float* feat, int C, const float* nxyz,
                      int N, int S, float r, int K, Prep* L,
                      float* dstF, int ldf, int coff) {
    off = mark;
    int rows = BB * S * K;
    int* idx = (int*)alloc((size_t)rows * 4);
    int q = BB * S;
    ball_query_kernel<<<dim3((q + 255) / 256), dim3(256), 0, stream>>>(
        xyz, nxyz, N, S, r * r, K, idx);
    int maxC = L[0].coutP; if (L[1].coutP > maxC) maxC = L[1].coutP;
    _Float16* Ya = (_Float16*)alloc((size_t)rows * maxC * 2);
    _Float16* Yb = (_Float16*)alloc((size_t)rows * maxC * 2);
    _Float16* Xg = (_Float16*)alloc((size_t)rows * L[0].cinP * 2);
    group_rows_kernel<<<dim3(rows), dim3(128), 0, stream>>>(
        feat, C, xyz, nxyz, idx, N, S, K, L[0].cinP, Xg);
    runMLP(L[0], Xg, rows, Ya, L[0].coutP, nullptr, 0, 1);
    runMLP(L[1], Ya, rows, Yb, L[1].coutP, nullptr, 0, 1);
    // last layer: GEMM with fused group-max-pool epilogue (no Y materialized)
    runMLPFull(L[2], Yb, rows, nullptr, 0, nullptr, 0, 1, dstF, ldf, coff, K);
  };

  // ---- SA level 1: 2048 -> 512, radii/nsample per scale, cin = 3 ----
  fps_kernel<<<dim3(BB), dim3(256), 0, stream>>>(points, 2048, 512, fps1);
  { int t = BB * 512 * 3;
    gather_xyz_kernel<<<dim3((t + 255) / 256), dim3(256), 0, stream>>>(points, fps1, 2048, 512, l1xyz); }
  { int n = BB * 512 * 448;
    zero_f32_kernel<<<dim3((n + 255) / 256), dim3(256), 0, stream>>>(l1pts, n); }
  { float r[3] = {0.1f, 0.2f, 0.4f}; int k[3] = {16, 32, 64}; int co[3] = {0, 64, 192};
    for (int i = 0; i < 3; ++i)
      runScale(points, nullptr, 0, l1xyz, 2048, 512, r[i], k[i], Pm1[i], l1pts, 448, co[i]); }

  // ---- SA level 2: 512 -> 256, cin = 448 + 3 ----
  fps_kernel<<<dim3(BB), dim3(256), 0, stream>>>(l1xyz, 512, 256, fps2);
  { int t = BB * 256 * 3;
    gather_xyz_kernel<<<dim3((t + 255) / 256), dim3(256), 0, stream>>>(l1xyz, fps2, 512, 256, l2xyz); }
  { int n = BB * 256 * 640;
    zero_f32_kernel<<<dim3((n + 255) / 256), dim3(256), 0, stream>>>(l2pts, n); }
  { float r[3] = {0.2f, 0.4f, 0.8f}; int k[3] = {32, 64, 128}; int co[3] = {0, 128, 384};
    for (int i = 0; i < 3; ++i)
      runScale(l1xyz, l1pts, 448, l2xyz, 512, 256, r[i], k[i], Pm2[i], l2pts, 640, co[i]); }

  // ---- SA group-all: rows = 8*256, [xyz|l2pts] -> 643 pad 672, max over 256 ----
  {
    off = mark;
    int rows = BB * 256;
    _Float16* Xg = (_Float16*)alloc((size_t)rows * Pssg[0].cinP * 2);
    _Float16* Ya = (_Float16*)alloc((size_t)rows * 512 * 2);
    _Float16* Yb = (_Float16*)alloc((size_t)rows * 512 * 2);
    concat2_f16_kernel<<<dim3(rows), dim3(128), 0, stream>>>(
        l2xyz, 3, l2pts, 640, 0, 256, Pssg[0].cinP, Xg);
    { int n = BB * 512;
      zero_f32_kernel<<<dim3((n + 255) / 256), dim3(256), 0, stream>>>(l3pts, n); }
    runMLP(Pssg[0], Xg, rows, Ya, Pssg[0].coutP, nullptr, 0, 1);
    runMLP(Pssg[1], Ya, rows, Yb, Pssg[1].coutP, nullptr, 0, 1);
    runMLPFull(Pssg[2], Yb, rows, nullptr, 0, nullptr, 0, 1, l3pts, 512, 0, 256);
    int tt = BB * 512;
    float* outTail = (float*)d_out + (size_t)BB * 2048 * 50;  // (B,512,1) == flat (B,512)
    copy_f32_kernel<<<dim3((tt + 255) / 256), dim3(256), 0, stream>>>(l3pts, outTail, tt);
  }

  // ---- FP3: broadcast l3 (S==1) + l2pts -> 1152 -> 256 -> 256 ----
  {
    off = mark;
    int rows = BB * 256;
    _Float16* Xg = (_Float16*)alloc((size_t)rows * 1152 * 2);
    _Float16* Ya = (_Float16*)alloc((size_t)rows * 256 * 2);
    concat2_f16_kernel<<<dim3(rows), dim3(128), 0, stream>>>(
        l2pts, 640, l3pts, 512, 1, 256, 1152, Xg);
    runMLP(Pfp3[0], Xg, rows, Ya, Pfp3[0].coutP, nullptr, 0, 1);
    runMLP(Pfp3[1], Ya, rows, nullptr, 0, l2new, 256, 1);
  }

  // ---- FP2: interp(l2new over l1xyz) + l1pts -> 704 -> 256 -> 128 ----
  {
    off = mark;
    int rows = BB * 512;
    float* interp = (float*)alloc((size_t)rows * 256 * 4);
    _Float16* Xg = (_Float16*)alloc((size_t)rows * 704 * 2);
    _Float16* Ya = (_Float16*)alloc((size_t)rows * 256 * 2);
    fp_interp_kernel<<<dim3((rows + 255) / 256), dim3(256), 0, stream>>>(
        l1xyz, l2xyz, l2new, 512, 256, 256, interp);
    concat2_f16_kernel<<<dim3(rows), dim3(128), 0, stream>>>(
        l1pts, 448, interp, 256, 0, 512, 704, Xg);
    runMLP(Pfp2[0], Xg, rows, Ya, Pfp2[0].coutP, nullptr, 0, 1);
    runMLP(Pfp2[1], Ya, rows, nullptr, 0, l1new, 128, 1);
  }

  // ---- FP1: [cls|xyz|xyz|interp(l1new)] -> 140 pad 160 -> 128 -> 128 ----
  {
    off = mark;
    int rows = BB * 2048;
    float* interp = (float*)alloc((size_t)rows * 128 * 4);
    _Float16* Xg = (_Float16*)alloc((size_t)rows * 160 * 2);
    _Float16* Ya = (_Float16*)alloc((size_t)rows * 128 * 2);
    fp_interp_kernel<<<dim3((rows + 255) / 256), dim3(256), 0, stream>>>(
        points, l1xyz, l1new, 2048, 512, 128, interp);
    build_fp1_kernel<<<dim3(rows), dim3(128), 0, stream>>>(cls, points, interp, 2048, 160, Xg);
    runMLP(Pfp1[0], Xg, rows, Ya, Pfp1[0].coutP, nullptr, 0, 1);
    runMLP(Pfp1[1], Ya, rows, l0feat, 128, nullptr, 0, 1);
  }

  // ---- Head: 128 -> 128 (BN+relu) -> 50 (pad 64) -> log_softmax ----
  {
    off = mark;
    int rows = BB * 2048;
    _Float16* feat = (_Float16*)alloc((size_t)rows * 128 * 2);
    runMLP(Ph1, l0feat, rows, feat, 128, nullptr, 0, 1);
    runMLP(Ph2, feat, rows, nullptr, 0, logits, 64, 0);
    log_softmax_kernel<<<dim3((rows + 255) / 256), dim3(256), 0, stream>>>(
        logits, 64, 50, rows, (float*)d_out);
  }
}
